// SelfAttentionBlock_38663295598640
// MI455X (gfx1250) — matrix operand order repaired and checked by reading, not compiled, not run on hardware
//
#include <hip/hip_runtime.h>
#include <hip/hip_bf16.h>

// ---------------------------------------------------------------------------
// Self-attention block (B=4, C=256, H=W=64 -> N=4096) for gfx1250 (MI455X).
// Flash-attention style fusion, bf16 WMMA (v_wmma_f32_16x16x32_bf16), fp32
// accumulation, online softmax in VGPRs. K/V tiles staged into LDS with
// gfx1250 async copies (GLOBAL_LOAD_ASYNC_TO_LDS_B128 + s_wait_asynccnt),
// double-buffered against the WMMA chain. Never materializes the N x N
// attention matrix (256 MB > 192 MB L2).
// ---------------------------------------------------------------------------

typedef __bf16 bf16;
typedef __attribute__((ext_vector_type(16))) __bf16 v16bf;
typedef __attribute__((ext_vector_type(8)))  __bf16 v8bf;
typedef __attribute__((ext_vector_type(8)))  float  v8f;
typedef __attribute__((ext_vector_type(4)))  int    v4i;

typedef v4i __attribute__((address_space(1))) g_v4i;  // global int4
typedef v4i __attribute__((address_space(3))) s_v4i;  // LDS int4

static constexpr int BB = 4;     // batch
static constexpr int CC = 256;   // channels
static constexpr int NN = 4096;  // tokens (H*W)

// ---- async global->LDS copy (ASYNCcnt path) --------------------------------
__device__ __forceinline__ void async_ld_b128(void* lds, const void* g) {
#if __has_builtin(__builtin_amdgcn_global_load_async_to_lds_b128)
  __builtin_amdgcn_global_load_async_to_lds_b128((g_v4i*)g, (s_v4i*)lds,
                                                 /*offset=*/0, /*cpol=*/0);
#else
  const unsigned lds_off = (unsigned)(uintptr_t)lds;  // low 32 bits = LDS addr
  asm volatile("global_load_async_to_lds_b128 %0, %1, off"
               :: "v"(lds_off), "v"(g) : "memory");
#endif
}

#if __has_builtin(__builtin_amdgcn_s_wait_asynccnt)
#define WAIT_ASYNC(n) __builtin_amdgcn_s_wait_asynccnt(n)
#else
#define WAIT_ASYNC(n) asm volatile("s_wait_asynccnt %0" :: "i"(n) : "memory")
#endif

// ---- WMMA helper -----------------------------------------------------------
__device__ __forceinline__ v8f wmma_bf16(v16bf a, v16bf b, v8f c) {
  // D = A(16x32 bf16) * B(32x16 bf16) + C(16x16 f32)
  return __builtin_amdgcn_wmma_f32_16x16x32_bf16(
      /*neg_a=*/false, a, /*neg_b=*/false, b,
      /*c_mod=*/(short)0, c, /*reuse_a=*/false, /*reuse_b=*/false);
}

// A-fragment (16x32, 16-bit). rowbase = this lane's row (contiguous K).
// lanes 0-15: elems 0..15 = K0..15
// lanes16-31: elems 0..15 = K16..31
__device__ __forceinline__ v16bf load_a16x32(const bf16* rowbase, int c0, int hi) {
  const v8bf lo  = *(const v8bf*)(rowbase + c0 + hi * 16);
  const v8bf hi8 = *(const v8bf*)(rowbase + c0 + hi * 16 + 8);
  return __builtin_shufflevector(lo, hi8, 0,1,2,3,4,5,6,7,8,9,10,11,12,13,14,15);
}

// ---- Kernel 1: transpose+convert x (f32 [b][c][n]) -> xT (bf16 [b][n][c]) --
__global__ void xpose_kernel(const float* __restrict__ x, bf16* __restrict__ xT) {
  __shared__ float tile[32][33];
  const int b  = blockIdx.z;
  const int n0 = blockIdx.x * 32;
  const int c0 = blockIdx.y * 32;
  const int tx = threadIdx.x;
#pragma unroll
  for (int i = threadIdx.y; i < 32; i += 8)
    tile[i][tx] = x[((size_t)b * CC + c0 + i) * NN + n0 + tx];
  __syncthreads();
#pragma unroll
  for (int i = threadIdx.y; i < 32; i += 8)
    xT[((size_t)b * NN + n0 + i) * CC + c0 + tx] = (bf16)tile[tx][i];
}

// ---- Kernel 2: convert weights f32 -> bf16 ---------------------------------
__global__ void convw_kernel(const float* __restrict__ Wq, const float* __restrict__ Wk,
                             const float* __restrict__ Wv, bf16* __restrict__ q,
                             bf16* __restrict__ k, bf16* __restrict__ v) {
  const int i = blockIdx.x * blockDim.x + threadIdx.x;
  if (i < CC * CC) {
    q[i] = (bf16)Wq[i];
    k[i] = (bf16)Wk[i];
    v[i] = (bf16)Wv[i];
  }
}

// ---- Kernel 3: QKV projection ----------------------------------------------
// q[b,o,n] = sum_c Wq[o,c] * x[b,c,n]; outputs: qT/kT bf16 [b][n][o], v bf16 [b][o][m].
__global__ __launch_bounds__(256, 1) void qkv_proj_kernel(
    const bf16* __restrict__ xT, const bf16* __restrict__ Wq,
    const bf16* __restrict__ Wk, const bf16* __restrict__ Wv,
    bf16* __restrict__ qT, bf16* __restrict__ kT, bf16* __restrict__ vv) {
  const int lane = threadIdx.x & 31;
  const int wv   = threadIdx.x >> 5;
  const int hi   = lane >> 4;
  const int l15  = lane & 15;

  const int wid = blockIdx.x * 8 + wv;  // 16384 waves: B * 16 o-tiles * 256 n-tiles
  const int b   = wid >> 12;
  const int t   = wid & 4095;
  const int o0  = (t >> 8) << 4;
  const int n0  = (t & 255) << 4;

  const bf16* xrow  = xT + ((size_t)b * NN + n0 + l15) * CC;   // B-frag source
  const bf16* wqrow = Wq + (size_t)(o0 + l15) * CC;            // A-frag sources
  const bf16* wkrow = Wk + (size_t)(o0 + l15) * CC;
  const bf16* wvrow = Wv + (size_t)(o0 + l15) * CC;

  v8f Dq{}, Dk{}, Dv{};
#pragma unroll
  for (int c8 = 0; c8 < 8; ++c8) {
    const int c0 = c8 * 32;
    const v16bf xb = *(const v16bf*)(xrow + c0 + hi * 16);
    Dq = wmma_bf16(load_a16x32(wqrow, c0, hi), xb, Dq);
    Dk = wmma_bf16(load_a16x32(wkrow, c0, hi), xb, Dk);
    Dv = wmma_bf16(load_a16x32(wvrow, c0, hi), xb, Dv);
  }

  // D layout: VGPR r holds row o = o0 + hi*8 + r, col n = n0 + l15.
  bf16* qdst = qT + ((size_t)b * NN + n0 + l15) * CC + o0 + hi * 8;
  bf16* kdst = kT + ((size_t)b * NN + n0 + l15) * CC + o0 + hi * 8;
  v8bf q8, k8;
#pragma unroll
  for (int r = 0; r < 8; ++r) { q8[r] = (bf16)Dq[r]; k8[r] = (bf16)Dk[r]; }
  *(v8bf*)qdst = q8;
  *(v8bf*)kdst = k8;
#pragma unroll
  for (int r = 0; r < 8; ++r)
    vv[((size_t)b * CC + o0 + hi * 8 + r) * NN + n0 + l15] = (bf16)Dv[r];
}

// ---- Kernel 4: fused flash attention + epilogue ----------------------------
// Block = 4 waves = 64 query rows; each wave: 16 query rows, full C=256 output
// in VGPRs. 32-key tiles staged to LDS with async copies, double-buffered and
// shared by all 4 waves. Grid = (N/64, B).
__global__ __launch_bounds__(128, 1) void attn_kernel(
    const bf16* __restrict__ qT, const bf16* __restrict__ kT,
    const bf16* __restrict__ vv, const float* __restrict__ x,
    const float* __restrict__ gamma, float* __restrict__ out) {
  const int tid  = threadIdx.x;
  const int lane = tid & 31;
  const int wv   = tid >> 5;
  const int hi   = lane >> 4;
  const int l15  = lane & 15;
  const int b    = blockIdx.y;
  const int n0   = blockIdx.x * 64 + wv * 16;

  __shared__ __align__(16) bf16 ktile[2][32][256];  // 2 x 16 KB: 32 keys x C
  __shared__ __align__(16) bf16 vtile[2][256][32];  // 2 x 16 KB: C x 32 keys
  __shared__ __align__(16) bf16 lds_p[4][16][32];   // per-wave P tile

  // ---- async staging of one 32-key K/V tile (16 x b128 per thread) ----
  auto stage = [&](int buf, int j) {
#pragma unroll
    for (int i = 0; i < 8; ++i) {
      const int ck = tid + i * 128;               // K chunk: 8 bf16 = 16 B
      const int krow = ck >> 5, kcol = (ck & 31) * 8;
      async_ld_b128(&ktile[buf][0][0] + ck * 8,
                    kT + ((size_t)b * NN + j + krow) * CC + kcol);
      const int cv = tid + i * 128;               // V chunk
      const int vrow = cv >> 2, vcol = (cv & 3) * 8;
      async_ld_b128(&vtile[buf][0][0] + cv * 8,
                    vv + ((size_t)b * CC + vrow) * NN + j + vcol);
    }
  };

  // Resident Q A-fragments: 16 rows x 256 channels (8 fragments of K=32).
  const bf16* qrow = qT + ((size_t)b * NN + n0 + l15) * CC;
  v16bf qA[8];
#pragma unroll
  for (int c8 = 0; c8 < 8; ++c8) qA[c8] = load_a16x32(qrow, c8 * 32, hi);

  v8f O[16];
#pragma unroll
  for (int i = 0; i < 16; ++i) O[i] = v8f{};
  float mrow[8], lrow[8];
#pragma unroll
  for (int r = 0; r < 8; ++r) { mrow[r] = -1e30f; lrow[r] = 0.0f; }

  stage(0, 0);  // prologue prefetch

  for (int j = 0; j < NN; j += 32) {
    const int cur = (j >> 5) & 1;
    if (j + 32 < NN) {
      stage(cur ^ 1, j + 32);   // prefetch next tile into alternate buffer
      WAIT_ASYNC(16);           // in-order ASYNCcnt: previous 16 are done
    } else {
      WAIT_ASYNC(0);
    }
    __syncthreads();            // all waves' stores to buf `cur` visible

    const bf16* kb  = &ktile[cur][0][0];
    const bf16* vbp = &vtile[cur][0][0];

    // --- S = Q * K^T for 32 keys (two 16x16 f32 fragments) ---
    v8f S0{}, S1{};
#pragma unroll
    for (int c8 = 0; c8 < 8; ++c8) {
      const int c0 = c8 * 32;
      const v16bf kb0 = *(const v16bf*)(kb + (size_t)l15 * CC + c0 + hi * 16);
      const v16bf kb1 = *(const v16bf*)(kb + (size_t)(16 + l15) * CC + c0 + hi * 16);
      S0 = wmma_bf16(qA[c8], kb0, S0);
      S1 = wmma_bf16(qA[c8], kb1, S1);
    }

    // --- online softmax (rows = r + hi*8; cols spread over 16-lane half) ---
    float scale[8], p0[8], p1[8];
#pragma unroll
    for (int r = 0; r < 8; ++r) {
      float mx = fmaxf(S0[r], S1[r]);
#pragma unroll
      for (int d = 1; d < 16; d <<= 1) mx = fmaxf(mx, __shfl_xor(mx, d, 32));
      const float mnew = fmaxf(mrow[r], mx);
      scale[r] = __expf(mrow[r] - mnew);
      p0[r] = __expf(S0[r] - mnew);
      p1[r] = __expf(S1[r] - mnew);
      float rs = p0[r] + p1[r];
#pragma unroll
      for (int d = 1; d < 16; d <<= 1) rs += __shfl_xor(rs, d, 32);
      lrow[r] = lrow[r] * scale[r] + rs;
      mrow[r] = mnew;
    }
#pragma unroll
    for (int cc = 0; cc < 16; ++cc)
#pragma unroll
      for (int r = 0; r < 8; ++r) O[cc][r] *= scale[r];

    // --- transpose P (D layout) -> A layout through LDS ---
#pragma unroll
    for (int r = 0; r < 8; ++r) {
      lds_p[wv][hi * 8 + r][l15]      = (bf16)p0[r];
      lds_p[wv][hi * 8 + r][16 + l15] = (bf16)p1[r];
    }
    asm volatile("s_wait_dscnt 0" ::: "memory");  // DS in-order within wave
    const v8bf plo = *(const v8bf*)&lds_p[wv][l15][hi * 16];
    const v8bf phi = *(const v8bf*)&lds_p[wv][l15][hi * 16 + 8];
    const v16bf pA = __builtin_shufflevector(plo, phi,
        0,1,2,3,4,5,6,7,8,9,10,11,12,13,14,15);

    // --- O += P * V (16 channel-column fragments) ---
#pragma unroll
    for (int cc = 0; cc < 16; ++cc) {
      const v16bf vb = *(const v16bf*)(vbp + (size_t)(cc * 16 + l15) * 32 + hi * 16);
      O[cc] = wmma_bf16(pA, vb, O[cc]);
    }
    __syncthreads();            // reads of buf `cur` done before it is restaged
  }

  // --- epilogue: out = gamma * (O / l) + x, transposed through LDS so each
  //     lane writes 64 B-contiguous runs along n. Reuse dead K/V buffers. ---
  const float g = *gamma;
  float invl[8];
#pragma unroll
  for (int r = 0; r < 8; ++r) invl[r] = 1.0f / lrow[r];

  __syncthreads();              // key loop fully done; K/V LDS is dead
  float* ldsO = (wv < 2) ? (float*)&ktile[wv][0][0]        // 16 KB per wave
                         : (float*)&vtile[wv - 2][0][0];
#pragma unroll
  for (int cc = 0; cc < 16; ++cc) {
    const int c = cc * 16 + l15;
#pragma unroll
    for (int r = 0; r < 8; ++r)
      ldsO[(r + hi * 8) * 256 + c] = g * O[cc][r] * invl[r];
  }
  asm volatile("s_wait_dscnt 0" ::: "memory");  // wave-internal transpose

  for (int ci = lane; ci < 256; ci += 32) {
    float vals[16];
#pragma unroll
    for (int r = 0; r < 16; ++r) vals[r] = ldsO[r * 256 + ci];
    const size_t gbase = ((size_t)b * CC + ci) * NN + n0;
#pragma unroll
    for (int r4 = 0; r4 < 4; ++r4) {
      const float4 xa = *(const float4*)(x + gbase + r4 * 4);
      float4 o;
      o.x = vals[r4 * 4 + 0] + xa.x;
      o.y = vals[r4 * 4 + 1] + xa.y;
      o.z = vals[r4 * 4 + 2] + xa.z;
      o.w = vals[r4 * 4 + 3] + xa.w;
      *(float4*)(out + gbase + r4 * 4) = o;
    }
  }
}

// ---------------------------------------------------------------------------
extern "C" void kernel_launch(void* const* d_in, const int* in_sizes, int n_in,
                              void* d_out, int out_size, void* d_ws, size_t ws_size,
                              hipStream_t stream) {
  const float* x     = (const float*)d_in[0];
  const float* Wq    = (const float*)d_in[1];
  const float* Wk    = (const float*)d_in[2];
  const float* Wv    = (const float*)d_in[3];
  const float* gamma = (const float*)d_in[4];
  float* out = (float*)d_out;

  // Workspace layout (bf16): xT, qT, kT, v, Wq, Wk, Wv  (~34 MB total)
  bf16* ws  = (bf16*)d_ws;
  const size_t tensor_elems = (size_t)BB * NN * CC;
  bf16* xT  = ws;
  bf16* qT  = xT + tensor_elems;
  bf16* kT  = qT + tensor_elems;
  bf16* vv  = kT + tensor_elems;
  bf16* Wqb = vv + tensor_elems;
  bf16* Wkb = Wqb + (size_t)CC * CC;
  bf16* Wvb = Wkb + (size_t)CC * CC;

  xpose_kernel<<<dim3(NN / 32, CC / 32, BB), dim3(32, 8), 0, stream>>>(x, xT);
  convw_kernel<<<(CC * CC + 255) / 256, 256, 0, stream>>>(Wq, Wk, Wv, Wqb, Wkb, Wvb);
  // 16384 waves (B * 16 o-tiles * 256 n-tiles), 8 waves per 256-thread block.
  qkv_proj_kernel<<<2048, 256, 0, stream>>>(xT, Wqb, Wkb, Wvb, qT, kT, vv);
  attn_kernel<<<dim3(NN / 64, BB), 128, 0, stream>>>(qT, kT, vv, x, gamma, out);
}
